// Encoder_25761213841458
// MI455X (gfx1250) — matrix operand order repaired and checked by reading, not compiled, not run on hardware
//
#include <hip/hip_runtime.h>
#include <hip/hip_bf16.h>
#include <stdint.h>

// ---------------------------------------------------------------------------
// Bidirectional transformer encoder block for MI455X (gfx1250, wave32, WMMA)
// All matmuls run on v_wmma_f32_16x16x32_bf16 (fp32 accumulate).
// GEMM A-tiles are staged with CDNA5 async global->LDS DMA (ASYNCcnt).
// ---------------------------------------------------------------------------

#define B_ 8
#define T_ 1024
#define C_ 1024
#define H_ 16
#define HS_ 64
#define BT_ (B_ * T_)      // 8192 rows of tokens
#define C2_ (2 * C_)       // 2048
#define C8_ (8 * C_)       // 8192

typedef __bf16 v16bf __attribute__((ext_vector_type(16)));
typedef float  v8f   __attribute__((ext_vector_type(8)));

__device__ __forceinline__ unsigned short f2bf(float f) {
  unsigned u = __float_as_uint(f);
  u += 0x7fffu + ((u >> 16) & 1u);               // round-to-nearest-even
  return (unsigned short)(u >> 16);
}

__device__ __forceinline__ v8f wmma_bf16(v16bf a, v16bf b, v8f c) {
  // D = A(16x32) * B(32x16) + C, fp32 accumulate
  return __builtin_amdgcn_wmma_f32_16x16x32_bf16(false, a, false, b,
                                                 (short)0, c, false, false);
}

// CDNA5 async DMA: copy 16 bytes global -> LDS, tracked by ASYNCcnt.
// Flat pointers to LDS carry the LDS byte offset in addr[31:0] (ISA 10.2),
// which is exactly what the instruction's VDST operand wants.
__device__ __forceinline__ void async_copy_b128(void* lds_dst,
                                                const void* gsrc) {
  unsigned lds_off = (unsigned)(unsigned long long)lds_dst;
  unsigned long long gaddr = (unsigned long long)gsrc;
  asm volatile("global_load_async_to_lds_b128 %0, %1, off"
               :: "v"(lds_off), "v"(gaddr) : "memory");
}

__device__ __forceinline__ void wait_async_all() {
  asm volatile("s_wait_asynccnt 0" ::: "memory");
}

// A-fragment (16x32 bf16, ISA 7.12.2): lane m = lane&15;
// lanes 0-15 hold K = {0..7, 16..23}, lanes 16-31 hold K = {8..15, 24..31}.
__device__ __forceinline__ v16bf load_a_frag(const unsigned short* base,
                                             int pitch, int row0, int lane) {
  int m  = row0 + (lane & 15);
  int kb = (lane < 16) ? 0 : 8;
  const unsigned short* p = base + m * pitch + kb;
  v16bf a;
  uint4* d = reinterpret_cast<uint4*>(&a);
  d[0] = *reinterpret_cast<const uint4*>(p);
  d[1] = *reinterpret_cast<const uint4*>(p + 16);
  return a;
}

// B-fragment (32x16 bf16) from K-major (transposed) LDS: row n holds all K.
// lane n = lane&15; lanes 0-15: K={0..7,16..23}, lanes 16-31: K={8..15,24..31}.
__device__ __forceinline__ v16bf load_b_frag(const unsigned short* base,
                                             int pitch, int col0, int lane) {
  int n  = col0 + (lane & 15);
  int kb = (lane < 16) ? 0 : 8;
  const unsigned short* p = base + n * pitch + kb;
  v16bf b;
  uint4* d = reinterpret_cast<uint4*>(&b);
  d[0] = *reinterpret_cast<const uint4*>(p);
  d[1] = *reinterpret_cast<const uint4*>(p + 16);
  return b;
}

// ---------------------------------------------------------------------------
// LayerNorm: one block per row. Optionally flips the row (backward branch).
// Emits fp32 (residual path) and bf16 (WMMA operand path).
// ---------------------------------------------------------------------------
__global__ __launch_bounds__(256)
void ln_kernel(const float* __restrict__ in, const float* __restrict__ g,
               const float* __restrict__ bt, float* __restrict__ outF,
               unsigned short* __restrict__ outB, int W, int flip) {
  __shared__ float s1[256];
  __shared__ float s2[256];
  const int row = blockIdx.x, tid = threadIdx.x;
  const float* xr = in + (size_t)row * W;
  float sum = 0.f, sq = 0.f;
  for (int c = tid; c < W; c += 256) {
    float x = xr[flip ? (W - 1 - c) : c];
    sum += x; sq += x * x;
  }
  s1[tid] = sum; s2[tid] = sq;
  __syncthreads();
  for (int off = 128; off > 0; off >>= 1) {
    if (tid < off) { s1[tid] += s1[tid + off]; s2[tid] += s2[tid + off]; }
    __syncthreads();
  }
  const float mu   = s1[0] / W;
  const float var  = s2[0] / W - mu * mu;
  const float rstd = rsqrtf(var + 1e-5f);
  for (int c = tid; c < W; c += 256) {
    float x = xr[flip ? (W - 1 - c) : c];
    float y = (x - mu) * rstd * g[c] + bt[c];
    if (outF) outF[(size_t)row * W + c] = y;
    if (outB) outB[(size_t)row * W + c] = f2bf(y);
  }
}

// fp32 -> bf16 elementwise
__global__ void cvt_plain(const float* __restrict__ in,
                          unsigned short* __restrict__ out, int n) {
  int i = blockIdx.x * 256 + threadIdx.x;
  if (i < n) out[i] = f2bf(in[i]);
}

// Wq/Wk/Wv [H,C,HS] -> bf16 [C, H*HS] so QKV is one big GEMM per projection
__global__ void cvt_qkv(const float* __restrict__ in,
                        unsigned short* __restrict__ out) {
  int o = blockIdx.x * 256 + threadIdx.x;          // o = c*1024 + h*64 + s
  if (o < C_ * C_) {
    int c = o >> 10, hs = o & 1023;
    int h = hs >> 6, s = hs & 63;
    out[o] = f2bf(in[((size_t)h * C_ + c) * HS_ + s]);
  }
}

// ---------------------------------------------------------------------------
// Generic bf16 GEMM: Out[M,N] = A[M,K] * Bw[K,N] (+bias)(+res)(ReLU?)
// Block tile 128x64, K-step 32, 8 waves -> each wave 32x32 (4 WMMA/K-step).
// Double-buffered LDS. A staged via global_load_async_to_lds_b128 (no VGPR
// round-trip, ASYNCcnt-tracked); B staged transposed for contiguous frag
// loads. M, N multiples of 128/64; K multiple of 32 (true for all uses).
// ---------------------------------------------------------------------------
#define LPITCH 40  // bf16 elements per LDS row (16B aligned, pads banks)

__device__ __forceinline__ void gemm_stage(
    unsigned short* As, unsigned short* Bs,
    const unsigned short* __restrict__ A, int lda,
    const unsigned short* __restrict__ Bw, int ldb,
    int row0, int col0, int kb, int tid) {
  const int k0 = kb * 32;
  // A tile 128x32: 512 chunks of 8 bf16, 2 async DMA transfers per thread
#pragma unroll
  for (int it = 0; it < 2; ++it) {
    int slot = tid + it * 256;
    int m = slot >> 2, kc = (slot & 3) * 8;
    async_copy_b128(As + m * LPITCH + kc,
                    A + (size_t)(row0 + m) * lda + k0 + kc);
  }
  // B tile 32x64 -> store transposed [n][k] (layout conversion needs VALU)
  {
    int kk = tid >> 3, nc = (tid & 7) * 8;
    unsigned short tmp[8];
    *reinterpret_cast<uint4*>(tmp) =
        *reinterpret_cast<const uint4*>(Bw + (size_t)(k0 + kk) * ldb + col0 + nc);
#pragma unroll
    for (int i = 0; i < 8; ++i) Bs[(nc + i) * LPITCH + kk] = tmp[i];
  }
}

__global__ __launch_bounds__(256)
void gemm_bf16(const unsigned short* __restrict__ A, int lda,
               const unsigned short* __restrict__ Bw, int ldb,
               const float* __restrict__ bias,
               const float* __restrict__ res, int ldres,
               float* __restrict__ outF, unsigned short* __restrict__ outB,
               int ldd, int K, int relu) {
  __shared__ unsigned short As[2][128 * LPITCH];
  __shared__ unsigned short Bs[2][64 * LPITCH];
  const int tid = threadIdx.x, lane = tid & 31, w = tid >> 5;
  const int row0 = blockIdx.y * 128, col0 = blockIdx.x * 64;
  const int mw = (w >> 1) * 32, nw = (w & 1) * 32;

  v8f acc00 = {0,0,0,0,0,0,0,0}, acc01 = acc00, acc10 = acc00, acc11 = acc00;
  const int nk = K >> 5;

  gemm_stage(As[0], Bs[0], A, lda, Bw, ldb, row0, col0, 0, tid);
  for (int kb = 0; kb < nk; ++kb) {
    wait_async_all();       // our async A-tile writes have landed in LDS
    __syncthreads();        // publish LDS (A async + B ds stores) to all waves
    const int cur = kb & 1;
    if (kb + 1 < nk) {
      gemm_stage(As[cur ^ 1], Bs[cur ^ 1], A, lda, Bw, ldb, row0, col0,
                 kb + 1, tid);
      if (kb + 2 < nk)   // warm L2 for the tile after next
        __builtin_prefetch(A + (size_t)(row0 + (tid >> 1)) * lda + (kb + 2) * 32,
                           0, 0);
    }
    v16bf a0 = load_a_frag(As[cur], LPITCH, mw, lane);
    v16bf a1 = load_a_frag(As[cur], LPITCH, mw + 16, lane);
    v16bf b0 = load_b_frag(Bs[cur], LPITCH, nw, lane);
    v16bf b1 = load_b_frag(Bs[cur], LPITCH, nw + 16, lane);
    acc00 = wmma_bf16(a0, b0, acc00);
    acc01 = wmma_bf16(a0, b1, acc01);
    acc10 = wmma_bf16(a1, b0, acc10);
    acc11 = wmma_bf16(a1, b1, acc11);
  }

  // Epilogue: C-layout (ISA 7.12.2): reg r -> row r + 8*(lane>=16), col lane&15
  const int l15 = lane & 15, half8 = (lane < 16) ? 0 : 8;
  v8f accs[2][2] = {{acc00, acc01}, {acc10, acc11}};
#pragma unroll
  for (int i = 0; i < 2; ++i)
#pragma unroll
    for (int j = 0; j < 2; ++j) {
      const int colg = col0 + nw + j * 16 + l15;
      const float bv = bias ? bias[colg] : 0.f;
#pragma unroll
      for (int r = 0; r < 8; ++r) {
        const int rowg = row0 + mw + i * 16 + half8 + r;
        float vv = accs[i][j][r] + bv;
        if (res)  vv += res[(size_t)rowg * ldres + colg];
        if (relu) vv = fmaxf(vv, 0.f);
        if (outF) outF[(size_t)rowg * ldd + colg] = vv;
        if (outB) outB[(size_t)rowg * ldd + colg] = f2bf(vv);
      }
    }
}

// ---------------------------------------------------------------------------
// Flash attention: grid (T/16, H, B), 1 wave (32 threads) per block.
// Online softmax over 32-key blocks; QK^T and PV both on WMMA.
// q/k/v/o layout: [(b*T + t)*1024 + h*64 + s] (bf16).
// Scale is 1/sqrt(C) = 1/32 exactly as in the reference.
// ---------------------------------------------------------------------------
__global__ __launch_bounds__(32)
void flash_attn(const unsigned short* __restrict__ q,
                const unsigned short* __restrict__ k,
                const unsigned short* __restrict__ v,
                unsigned short* __restrict__ o) {
  __shared__ unsigned short pLds[16 * LPITCH];   // P tile 16x32 (bf16)
  __shared__ unsigned short vLds[64 * LPITCH];   // V^T tile [s][key]
  const int lane  = threadIdx.x;
  const int mtile = blockIdx.x, h = blockIdx.y, b = blockIdx.z;
  const size_t rowbase = ((size_t)b * T_) * C_ + h * HS_;  // + t*1024 + s
  const int l15 = lane & 15, half8 = (lane < 16) ? 0 : 8;
  const float sc = 0.03125f;                      // 1/sqrt(1024)

  // Q A-fragments (K-dim 64 -> two 16x32 frags)
  v16bf qa[2];
  {
    const int t  = mtile * 16 + l15;
    const int kb = (lane < 16) ? 0 : 8;
    const unsigned short* p = q + rowbase + (size_t)t * C_ + kb;
#pragma unroll
    for (int qt = 0; qt < 2; ++qt) {
      uint4* d = reinterpret_cast<uint4*>(&qa[qt]);
      d[0] = *reinterpret_cast<const uint4*>(p + qt * 32);
      d[1] = *reinterpret_cast<const uint4*>(p + qt * 32 + 16);
    }
  }

  float mi[8], li[8];
  v8f acc[4];
#pragma unroll
  for (int r = 0; r < 8; ++r) { mi[r] = -1e30f; li[r] = 0.f; }
#pragma unroll
  for (int st = 0; st < 4; ++st) acc[st] = (v8f){0,0,0,0,0,0,0,0};

  for (int jb = 0; jb < T_ / 32; ++jb) {
    const int kblock = jb * 32;

    // S = Q * K^T for 32 keys: two 16x16 tiles, contraction 64 (2 frags each)
    v8f s0 = {0,0,0,0,0,0,0,0}, s1 = s0;
#pragma unroll
    for (int ct = 0; ct < 2; ++ct) {
      const int cb = ct * 32 + ((lane < 16) ? 0 : 8);
#pragma unroll
      for (int st = 0; st < 2; ++st) {
        const int key = kblock + st * 16 + l15;
        const unsigned short* p = k + rowbase + (size_t)key * C_ + cb;
        v16bf kf;
        uint4* d = reinterpret_cast<uint4*>(&kf);
        d[0] = *reinterpret_cast<const uint4*>(p);
        d[1] = *reinterpret_cast<const uint4*>(p + 16);
        if (st == 0) s0 = wmma_bf16(qa[ct], kf, s0);
        else         s1 = wmma_bf16(qa[ct], kf, s1);
      }
    }

    __syncthreads();  // WAR: previous iter's LDS reads done before restage

    // Stage V block (32 keys x dims 0..31) transposed into LDS: vLds[s][key]
    {
      const unsigned short* vp = v + rowbase + (size_t)(kblock + lane) * C_;
#pragma unroll
      for (int c = 0; c < 4; ++c) {
        unsigned short tmp[8];
        *reinterpret_cast<uint4*>(tmp) =
            *reinterpret_cast<const uint4*>(vp + c * 8);
#pragma unroll
        for (int i = 0; i < 8; ++i) vLds[(c * 8 + i) * LPITCH + lane] = tmp[i];
      }
    }

    // Online softmax (row stats replicated across the 16 lanes of each half)
    float alpha[8];
#pragma unroll
    for (int r = 0; r < 8; ++r) {
      float v0 = s0[r] * sc, v1 = s1[r] * sc;
      float mx = fmaxf(v0, v1);
      mx = fmaxf(mx, __shfl_xor(mx, 1));
      mx = fmaxf(mx, __shfl_xor(mx, 2));
      mx = fmaxf(mx, __shfl_xor(mx, 4));
      mx = fmaxf(mx, __shfl_xor(mx, 8));
      const float mn = fmaxf(mi[r], mx);
      alpha[r] = __expf(mi[r] - mn);
      mi[r] = mn;
      const float p0 = __expf(v0 - mn), p1 = __expf(v1 - mn);
      float rs = p0 + p1;
      rs += __shfl_xor(rs, 1);
      rs += __shfl_xor(rs, 2);
      rs += __shfl_xor(rs, 4);
      rs += __shfl_xor(rs, 8);
      li[r] = li[r] * alpha[r] + rs;
      pLds[(r + half8) * LPITCH + l15]      = f2bf(p0);
      pLds[(r + half8) * LPITCH + 16 + l15] = f2bf(p1);
    }
#pragma unroll
    for (int st = 0; st < 4; ++st)
#pragma unroll
      for (int r = 0; r < 8; ++r) acc[st][r] *= alpha[r];

    __syncthreads();  // RAW: P and V^T staged before fragment loads

    // O += P(16x32) * V(32x64): 4 WMMAs
    const v16bf pa = load_a_frag(pLds, LPITCH, 0, lane);
#pragma unroll
    for (int st = 0; st < 4; ++st) {
      const v16bf vb = load_b_frag(vLds, LPITCH, st * 16, lane);
      acc[st] = wmma_bf16(pa, vb, acc[st]);
    }
  }

  // Normalize and store bf16
#pragma unroll
  for (int st = 0; st < 4; ++st)
#pragma unroll
    for (int r = 0; r < 8; ++r) {
      const float val = acc[st][r] / li[r];
      const int t = mtile * 16 + r + half8;
      o[rowbase + (size_t)t * C_ + st * 16 + l15] = f2bf(val);
    }
}

// ---------------------------------------------------------------------------
// Host orchestration
// ---------------------------------------------------------------------------
extern "C" void kernel_launch(void* const* d_in, const int* in_sizes, int n_in,
                              void* d_out, int out_size, void* d_ws,
                              size_t ws_size, hipStream_t stream) {
  (void)in_sizes; (void)n_in; (void)out_size; (void)ws_size;

  const float* x = (const float*)d_in[0];
  const float* Wq[2] = {(const float*)d_in[1],  (const float*)d_in[9]};
  const float* bq[2] = {(const float*)d_in[2],  (const float*)d_in[10]};
  const float* Wk[2] = {(const float*)d_in[3],  (const float*)d_in[11]};
  const float* bk[2] = {(const float*)d_in[4],  (const float*)d_in[12]};
  const float* Wv[2] = {(const float*)d_in[5],  (const float*)d_in[13]};
  const float* bv[2] = {(const float*)d_in[6],  (const float*)d_in[14]};
  const float* Wp[2] = {(const float*)d_in[7],  (const float*)d_in[15]};
  const float* bp[2] = {(const float*)d_in[8],  (const float*)d_in[16]};
  const float* lng[2] = {(const float*)d_in[17], (const float*)d_in[19]};
  const float* lnb[2] = {(const float*)d_in[18], (const float*)d_in[20]};
  const float* ln2g = (const float*)d_in[21];
  const float* ln2b = (const float*)d_in[22];
  const float* W1 = (const float*)d_in[23];
  const float* b1 = (const float*)d_in[24];
  const float* W2 = (const float*)d_in[25];
  const float* b2 = (const float*)d_in[26];
  float* out = (float*)d_out;

  // Workspace bump allocator (~472 MiB total)
  char* ws = (char*)d_ws;
  size_t off = 0;
  auto alloc = [&](size_t bytes) -> void* {
    void* p = ws + off;
    off += (bytes + 255) & ~(size_t)255;
    return p;
  };
  float*          h    = (float*)alloc((size_t)BT_ * C2_ * 4);   // concat out
  float*          hln  = (float*)alloc((size_t)BT_ * C2_ * 4);   // LN2 f32
  unsigned short* hlnb = (unsigned short*)alloc((size_t)BT_ * C2_ * 2);
  float*          lnf  = (float*)alloc((size_t)BT_ * C_ * 4);    // LN1 f32
  unsigned short* lnbf = (unsigned short*)alloc((size_t)BT_ * C_ * 2);
  unsigned short* qbuf = (unsigned short*)alloc((size_t)BT_ * C_ * 2);
  unsigned short* kbuf = (unsigned short*)alloc((size_t)BT_ * C_ * 2);
  unsigned short* vbuf = (unsigned short*)alloc((size_t)BT_ * C_ * 2);
  unsigned short* abuf = (unsigned short*)alloc((size_t)BT_ * C_ * 2);
  unsigned short* wqb  = (unsigned short*)alloc((size_t)C_ * C_ * 2);
  unsigned short* wkb  = (unsigned short*)alloc((size_t)C_ * C_ * 2);
  unsigned short* wvb  = (unsigned short*)alloc((size_t)C_ * C_ * 2);
  unsigned short* wpb  = (unsigned short*)alloc((size_t)C_ * C_ * 2);
  unsigned short* w1b  = (unsigned short*)alloc((size_t)C2_ * C8_ * 2);
  unsigned short* w2b  = (unsigned short*)alloc((size_t)C8_ * C2_ * 2);
  unsigned short* f1   = (unsigned short*)alloc((size_t)BT_ * C8_ * 2);

  // FFN weights -> bf16 (16.7M elements each)
  cvt_plain<<<(C2_ * C8_ + 255) / 256, 256, 0, stream>>>(W1, w1b, C2_ * C8_);
  cvt_plain<<<(C8_ * C2_ + 255) / 256, 256, 0, stream>>>(W2, w2b, C8_ * C2_);

  const dim3 gQKV(C_ / 64, BT_ / 128);       // N=1024
  for (int br = 0; br < 2; ++br) {
    // LN(flip?) -> fp32 residual + bf16 operand
    ln_kernel<<<BT_, 256, 0, stream>>>(x, lng[br], lnb[br], lnf, lnbf, C_, br);
    // Per-branch weights -> bf16 layouts
    cvt_qkv<<<(C_ * C_ + 255) / 256, 256, 0, stream>>>(Wq[br], wqb);
    cvt_qkv<<<(C_ * C_ + 255) / 256, 256, 0, stream>>>(Wk[br], wkb);
    cvt_qkv<<<(C_ * C_ + 255) / 256, 256, 0, stream>>>(Wv[br], wvb);
    cvt_plain<<<(C_ * C_ + 255) / 256, 256, 0, stream>>>(Wp[br], wpb, C_ * C_);
    // QKV projections
    gemm_bf16<<<gQKV, 256, 0, stream>>>(lnbf, C_, wqb, C_, bq[br],
                                        nullptr, 0, nullptr, qbuf, C_, C_, 0);
    gemm_bf16<<<gQKV, 256, 0, stream>>>(lnbf, C_, wkb, C_, bk[br],
                                        nullptr, 0, nullptr, kbuf, C_, C_, 0);
    gemm_bf16<<<gQKV, 256, 0, stream>>>(lnbf, C_, wvb, C_, bv[br],
                                        nullptr, 0, nullptr, vbuf, C_, C_, 0);
    // Attention
    flash_attn<<<dim3(T_ / 16, H_, B_), 32, 0, stream>>>(qbuf, kbuf, vbuf, abuf);
    // Output projection + residual(ln x), written into its half of concat h
    gemm_bf16<<<gQKV, 256, 0, stream>>>(abuf, C_, wpb, C_, bp[br],
                                        lnf, C_, h + br * C_, nullptr, C2_,
                                        C_, 0);
  }

  // LN2 over concat -> fp32 (final residual) + bf16 (FFN operand)
  ln_kernel<<<BT_, 256, 0, stream>>>(h, ln2g, ln2b, hln, hlnb, C2_, 0);
  // FFN1: relu(h_ln @ W1 + b1) -> bf16
  gemm_bf16<<<dim3(C8_ / 64, BT_ / 128), 256, 0, stream>>>(
      hlnb, C2_, w1b, C8_, b1, nullptr, 0, nullptr, f1, C8_, C2_, 1);
  // FFN2: f1 @ W2 + b2 + h_ln -> d_out (fp32)
  gemm_bf16<<<dim3(C2_ / 64, BT_ / 128), 256, 0, stream>>>(
      f1, C8_, w2b, C2_, b2, hln, C2_, out, nullptr, C2_, C8_, 0);
}
